// GraphNetBlock_40346922778973
// MI455X (gfx1250) — matrix operand order repaired and checked by reading, not compiled, not run on hardware
//
#include <hip/hip_runtime.h>

#define H 128
#define NNODES 100000
#define NEDGES 500000
#define ETILES (NEDGES/16)   /* 31250 */
#define NTILES (NNODES/16)   /* 6250  */

/* fragment table offsets (units of 512-ushort fragments), layout [kt][nt], nt=8 */
#define EW1F 0      /* 12 kt x 8 nt = 96 */
#define EW2F 96     /*  4 kt x 8 nt = 32 */
#define PW1F 128    /*  8 kt x 8 nt = 64 */
#define PW2F 192    /* 32 */
#define GW1F 224    /* 64 */
#define GW2F 288    /* 32 */
#define BW1F 320    /* 32 */
#define BW2F 352    /* 32 */
#define NFRAGS 384

typedef __attribute__((ext_vector_type(16))) __bf16        v16bf;
typedef __attribute__((ext_vector_type(2)))  __bf16        v2bf;
typedef __attribute__((ext_vector_type(8)))  float         v8f;
typedef __attribute__((ext_vector_type(8)))  unsigned int  v8u;

/* single fp32 -> bf16, round-to-nearest (ties away): 2 VALU */
__device__ __forceinline__ unsigned short f2bf(float f) {
  return (unsigned short)((__float_as_uint(f) + 0x8000u) >> 16);
}

/* pack 2 fp32 -> 2 bf16 in one dword.
   Preferred: hardware V_CVT_PK_BF16_F32 (if the toolchain declares it).
   Fallback: 2x round-bias add + 1x v_perm_b32 to pack the high halves (3 VALU). */
__device__ __forceinline__ unsigned pk2(float a, float b) {
#if __has_builtin(__builtin_amdgcn_cvt_pk_bf16_f32)
  v2bf r = __builtin_amdgcn_cvt_pk_bf16_f32(a, b);
  return __builtin_bit_cast(unsigned, r);
#else
  unsigned ua = __float_as_uint(a) + 0x8000u;
  unsigned ub = __float_as_uint(b) + 0x8000u;
  /* dest bytes: [0]=a.b2 [1]=a.b3 [2]=b.b2 [3]=b.b3  (s1 bytes=0..3, s0 bytes=4..7) */
  return __builtin_amdgcn_perm(ub, ua, 0x07060302u);
#endif
}

__device__ __forceinline__ v8f wmma_bf16(v16bf a, v16bf b, v8f c) {
  return __builtin_amdgcn_wmma_f32_16x16x32_bf16(false, a, false, b, (short)0, c,
                                                 false, false);
}

__device__ __forceinline__ void ds_fence() {
  asm volatile("s_wait_dscnt 0x0" ::: "memory");
}

__device__ __forceinline__ void atomic_fadd(float* p, float v) {
  unsigned long long ap = (unsigned long long)p;
  asm volatile("global_atomic_add_f32 %0, %1, off" :: "v"(ap), "v"(v) : "memory");
}

/* A fragment (16x32 bf16, MxK): lane row m=lane&15; elem i -> k = (i>>3)*16 + half*8 + (i&7).
   Built from an fp32 row in global memory (optionally scaled). */
__device__ __forceinline__ v16bf a_frag_f32(const float* __restrict__ row, int koff,
                                            int half, float scale) {
  const float4* p0 = (const float4*)(row + koff + half * 8);
  const float4* p1 = (const float4*)(row + koff + 16 + half * 8);
  float4 a = p0[0], b = p0[1], c = p1[0], d = p1[1];
  v8u u;
  u[0] = pk2(a.x * scale, a.y * scale); u[1] = pk2(a.z * scale, a.w * scale);
  u[2] = pk2(b.x * scale, b.y * scale); u[3] = pk2(b.z * scale, b.w * scale);
  u[4] = pk2(c.x * scale, c.y * scale); u[5] = pk2(c.z * scale, c.w * scale);
  u[6] = pk2(d.x * scale, d.y * scale); u[7] = pk2(d.z * scale, d.w * scale);
  return __builtin_bit_cast(v16bf, u);
}

/* A fragment from a row-major [16][128] bf16 tile staged in LDS. */
__device__ __forceinline__ v16bf a_frag_lds(const unsigned short* hid, int m,
                                            int koff, int half) {
  const uint4* p0 = (const uint4*)(hid + m * H + koff + half * 8);
  const uint4* p1 = (const uint4*)(hid + m * H + koff + 16 + half * 8);
  uint4 x = *p0, y = *p1;
  v8u u = {x.x, x.y, x.z, x.w, y.x, y.y, y.z, y.w};
  return __builtin_bit_cast(v16bf, u);
}

/* B fragment: pre-transformed layout frags[f][lane][16 bf16] -> 32B contiguous/lane. */
__device__ __forceinline__ v16bf b_frag(const unsigned short* __restrict__ frags,
                                        int f, int lane) {
  const uint4* p = (const uint4*)(frags + (long)f * 512 + lane * 16);
  uint4 x = p[0], y = p[1];
  v8u u = {x.x, x.y, x.z, x.w, y.x, y.y, y.z, y.w};
  return __builtin_bit_cast(v16bf, u);
}

/* Store one C tile (bias, optional relu) into the LDS staging tile as bf16,
   converting row-pairs with one packed convert each. */
__device__ __forceinline__ void c_to_lds(unsigned short* hid, const v8f* acc,
                                         const float* __restrict__ bias,
                                         int half, int mloc, bool relu) {
#pragma unroll
  for (int nt = 0; nt < 8; ++nt) {
    float b = bias[nt * 16 + mloc];
    int n = nt * 16 + mloc;
#pragma unroll
    for (int v = 0; v < 8; v += 2) {
      float va = acc[nt][v] + b;
      float vb = acc[nt][v + 1] + b;
      if (relu) { va = fmaxf(va, 0.f); vb = fmaxf(vb, 0.f); }
      unsigned p = pk2(va, vb);
      hid[(half * 8 + v) * H + n]     = (unsigned short)(p & 0xFFFFu);
      hid[(half * 8 + v + 1) * H + n] = (unsigned short)(p >> 16);
    }
  }
}

/* ---------------- kernel 0: zero sums (= x_new region of d_out) + counts -------- */
__global__ __launch_bounds__(256) void zero_kernel(float* __restrict__ sums,
                                                   float* __restrict__ counts) {
  int tid = blockIdx.x * 256 + threadIdx.x;
  float4 z = {0.f, 0.f, 0.f, 0.f};
  if (tid < (NNODES * H) / 4) ((float4*)sums)[tid] = z;
  if (tid < NNODES / 4) ((float4*)counts)[tid] = z;
}

/* ---------------- kernel 1: weights -> bf16 WMMA B-fragment layout -------------- */
__global__ __launch_bounds__(256) void wtrans_kernel(
    const float* __restrict__ ew1, const float* __restrict__ ew2,
    const float* __restrict__ pw1, const float* __restrict__ pw2,
    const float* __restrict__ gw1, const float* __restrict__ gw2,
    const float* __restrict__ bw1, const float* __restrict__ bw2,
    unsigned short* __restrict__ frags) {
  int tid = blockIdx.x * 256 + threadIdx.x;
  if (tid >= NFRAGS * 512) return;
  int f = tid >> 9;
  int l = (tid >> 4) & 31;
  int j = tid & 15;
  const float* W; int base;
  if (f < EW2F)      { W = ew1; base = EW1F; }
  else if (f < PW1F) { W = ew2; base = EW2F; }
  else if (f < PW2F) { W = pw1; base = PW1F; }
  else if (f < GW1F) { W = pw2; base = PW2F; }
  else if (f < GW2F) { W = gw1; base = GW1F; }
  else if (f < BW1F) { W = gw2; base = GW2F; }
  else if (f < BW2F) { W = bw1; base = BW1F; }
  else               { W = bw2; base = BW2F; }
  int local = f - base;
  int kt = local >> 3, nt = local & 7;
  /* B elem j of lane l: k = kt*32 + (j>>3)*16 + (l>>4)*8 + (j&7) ; n = nt*16 + (l&15) */
  int k = kt * 32 + ((j >> 3) << 4) + ((l >> 4) << 3) + (j & 7);
  int n = nt * 16 + (l & 15);
  frags[tid] = f2bf(W[(long)k * H + n]);
}

/* ---------------- kernel 2: edge pipeline (edge MLP + message MLP + scatter) ---- */
__global__ __launch_bounds__(256) void edge_kernel(
    const float* __restrict__ x, const int* __restrict__ edge_index,
    const float* __restrict__ edge_attr,
    const float* __restrict__ eb1, const float* __restrict__ eb2,
    const float* __restrict__ pb1, const float* __restrict__ pb2,
    const unsigned short* __restrict__ frags,
    float* __restrict__ sums, float* __restrict__ counts,
    float* __restrict__ edge_out) {
  __shared__ __align__(16) unsigned short hidall[8 * 16 * H];
  int wave = blockIdx.x * 8 + (threadIdx.x >> 5);
  if (wave >= ETILES) return;
  unsigned short* hid = hidall + (threadIdx.x >> 5) * 16 * H;

  int lane = threadIdx.x & 31;
  int half = lane >> 4;
  int mloc = lane & 15;          /* A-row / C-column index */
  int e    = wave * 16 + mloc;
  int r    = edge_index[e];
  int c    = edge_index[NEDGES + e];
  const float* xr = x + (long)r * H;
  const float* xc = x + (long)c * H;
  const float* ea = edge_attr + (long)e * H;

  /* C-rows owned by this lane are m = half*8 + v ; their dest (row) indices: */
  int rv[8];
#pragma unroll
  for (int v = 0; v < 8; ++v) rv[v] = edge_index[wave * 16 + half * 8 + v];

  v8f acc[8];
  v8f zero8 = {0.f, 0.f, 0.f, 0.f, 0.f, 0.f, 0.f, 0.f};

  /* ---- edge MLP layer1: [16,384] @ ew1 ---- */
#pragma unroll
  for (int nt = 0; nt < 8; ++nt) acc[nt] = zero8;
#pragma unroll
  for (int kt = 0; kt < 12; ++kt) {
    const float* src; int ko;
    if (kt < 4)       { src = xr; ko = kt * 32; }
    else if (kt < 8)  { src = xc; ko = (kt - 4) * 32; }
    else              { src = ea; ko = (kt - 8) * 32; }
    v16bf a = a_frag_f32(src, ko, half, 1.0f);
#pragma unroll
    for (int nt = 0; nt < 8; ++nt)
      acc[nt] = wmma_bf16(a, b_frag(frags, EW1F + kt * 8 + nt, lane), acc[nt]);
  }
  c_to_lds(hid, acc, eb1, half, mloc, true);
  ds_fence();

  /* ---- edge MLP layer2: hidden @ ew2 -> new_edge_attr ---- */
#pragma unroll
  for (int nt = 0; nt < 8; ++nt) acc[nt] = zero8;
#pragma unroll
  for (int kt = 0; kt < 4; ++kt) {
    v16bf a = a_frag_lds(hid, mloc, kt * 32, half);
#pragma unroll
    for (int nt = 0; nt < 8; ++nt)
      acc[nt] = wmma_bf16(a, b_frag(frags, EW2F + kt * 8 + nt, lane), acc[nt]);
  }
  ds_fence();
  /* bias -> global fp32 out, and stage bf16 copy for the message MLP */
#pragma unroll
  for (int nt = 0; nt < 8; ++nt) {
    float b = eb2[nt * 16 + mloc];
    int n = nt * 16 + mloc;
#pragma unroll
    for (int v = 0; v < 8; v += 2) {
      float va = acc[nt][v] + b;
      float vb = acc[nt][v + 1] + b;
      edge_out[(long)(wave * 16 + half * 8 + v) * H + n]     = va;
      edge_out[(long)(wave * 16 + half * 8 + v + 1) * H + n] = vb;
      unsigned p = pk2(va, vb);
      hid[(half * 8 + v) * H + n]     = (unsigned short)(p & 0xFFFFu);
      hid[(half * 8 + v + 1) * H + n] = (unsigned short)(p >> 16);
    }
  }
  ds_fence();

  /* ---- message MLP layer1: [x_col | new_edge] @ pw1 ---- */
#pragma unroll
  for (int nt = 0; nt < 8; ++nt) acc[nt] = zero8;
#pragma unroll
  for (int kt = 0; kt < 8; ++kt) {
    v16bf a = (kt < 4) ? a_frag_f32(xc, kt * 32, half, 1.0f)
                       : a_frag_lds(hid, mloc, (kt - 4) * 32, half);
#pragma unroll
    for (int nt = 0; nt < 8; ++nt)
      acc[nt] = wmma_bf16(a, b_frag(frags, PW1F + kt * 8 + nt, lane), acc[nt]);
  }
  ds_fence();
  c_to_lds(hid, acc, pb1, half, mloc, true);
  ds_fence();

  /* ---- message MLP layer2 + scatter-add into sums/counts ---- */
#pragma unroll
  for (int nt = 0; nt < 8; ++nt) acc[nt] = zero8;
#pragma unroll
  for (int kt = 0; kt < 4; ++kt) {
    v16bf a = a_frag_lds(hid, mloc, kt * 32, half);
#pragma unroll
    for (int nt = 0; nt < 8; ++nt)
      acc[nt] = wmma_bf16(a, b_frag(frags, PW2F + kt * 8 + nt, lane), acc[nt]);
  }
#pragma unroll
  for (int nt = 0; nt < 8; ++nt) {
    float b = pb2[nt * 16 + mloc];
#pragma unroll
    for (int v = 0; v < 8; ++v)
      atomic_fadd(sums + (long)rv[v] * H + nt * 16 + mloc, acc[nt][v] + b);
  }
  if (lane < 16) atomic_fadd(counts + r, 1.0f);
}

/* ---------------- kernel 3: node pipeline (gamma MLP + beta MLP) ---------------- */
__global__ __launch_bounds__(256) void node_kernel(
    const float* __restrict__ x, const float* __restrict__ counts,
    const float* __restrict__ gb1, const float* __restrict__ gb2,
    const float* __restrict__ bb1, const float* __restrict__ bb2,
    const unsigned short* __restrict__ frags,
    float* __restrict__ sums_xnew /* sums in, x_new out (same region) */) {
  __shared__ __align__(16) unsigned short hidall[8 * 16 * H];
  int wave = blockIdx.x * 8 + (threadIdx.x >> 5);
  if (wave >= NTILES) return;
  unsigned short* hid = hidall + (threadIdx.x >> 5) * 16 * H;

  int lane = threadIdx.x & 31;
  int half = lane >> 4;
  int mloc = lane & 15;
  int nd   = wave * 16 + mloc;
  const float* xrow = x + (long)nd * H;
  const float* srow = sums_xnew + (long)nd * H;
  float inv = 1.0f / fmaxf(counts[nd], 1.0f);

  v8f acc[8], x1[8];
  v8f zero8 = {0.f, 0.f, 0.f, 0.f, 0.f, 0.f, 0.f, 0.f};

  /* ---- gamma layer1: [x | agg] @ gw1 (agg = sums*inv folded into A conversion) */
#pragma unroll
  for (int nt = 0; nt < 8; ++nt) acc[nt] = zero8;
#pragma unroll
  for (int kt = 0; kt < 8; ++kt) {
    v16bf a = (kt < 4) ? a_frag_f32(xrow, kt * 32, half, 1.0f)
                       : a_frag_f32(srow, (kt - 4) * 32, half, inv);
#pragma unroll
    for (int nt = 0; nt < 8; ++nt)
      acc[nt] = wmma_bf16(a, b_frag(frags, GW1F + kt * 8 + nt, lane), acc[nt]);
  }
  c_to_lds(hid, acc, gb1, half, mloc, true);
  ds_fence();

  /* ---- gamma layer2 -> x1 (keep fp32) ---- */
#pragma unroll
  for (int nt = 0; nt < 8; ++nt) x1[nt] = zero8;
#pragma unroll
  for (int kt = 0; kt < 4; ++kt) {
    v16bf a = a_frag_lds(hid, mloc, kt * 32, half);
#pragma unroll
    for (int nt = 0; nt < 8; ++nt)
      x1[nt] = wmma_bf16(a, b_frag(frags, GW2F + kt * 8 + nt, lane), x1[nt]);
  }
  ds_fence();
#pragma unroll
  for (int nt = 0; nt < 8; ++nt) {
    float b = gb2[nt * 16 + mloc];
    int n = nt * 16 + mloc;
#pragma unroll
    for (int v = 0; v < 8; v += 2) {
      x1[nt][v] += b;
      x1[nt][v + 1] += b;
      unsigned p = pk2(x1[nt][v], x1[nt][v + 1]);   /* beta input (no relu) */
      hid[(half * 8 + v) * H + n]     = (unsigned short)(p & 0xFFFFu);
      hid[(half * 8 + v + 1) * H + n] = (unsigned short)(p >> 16);
    }
  }
  ds_fence();

  /* ---- beta layer1 ---- */
#pragma unroll
  for (int nt = 0; nt < 8; ++nt) acc[nt] = zero8;
#pragma unroll
  for (int kt = 0; kt < 4; ++kt) {
    v16bf a = a_frag_lds(hid, mloc, kt * 32, half);
#pragma unroll
    for (int nt = 0; nt < 8; ++nt)
      acc[nt] = wmma_bf16(a, b_frag(frags, BW1F + kt * 8 + nt, lane), acc[nt]);
  }
  ds_fence();
  c_to_lds(hid, acc, bb1, half, mloc, true);
  ds_fence();

  /* ---- beta layer2, residual add, final store ---- */
#pragma unroll
  for (int nt = 0; nt < 8; ++nt) acc[nt] = zero8;
#pragma unroll
  for (int kt = 0; kt < 4; ++kt) {
    v16bf a = a_frag_lds(hid, mloc, kt * 32, half);
#pragma unroll
    for (int nt = 0; nt < 8; ++nt)
      acc[nt] = wmma_bf16(a, b_frag(frags, BW2F + kt * 8 + nt, lane), acc[nt]);
  }
#pragma unroll
  for (int nt = 0; nt < 8; ++nt) {
    float b = bb2[nt * 16 + mloc];
#pragma unroll
    for (int v = 0; v < 8; ++v)
      sums_xnew[(long)(wave * 16 + half * 8 + v) * H + nt * 16 + mloc] =
          x1[nt][v] + acc[nt][v] + b;
  }
}

extern "C" void kernel_launch(void* const* d_in, const int* in_sizes, int n_in,
                              void* d_out, int out_size, void* d_ws, size_t ws_size,
                              hipStream_t stream) {
  (void)in_sizes; (void)n_in; (void)out_size; (void)ws_size;
  const float* x   = (const float*)d_in[0];
  const int*   ei  = (const int*)d_in[1];
  const float* ea  = (const float*)d_in[2];
  const float* ew1 = (const float*)d_in[3];  const float* eb1 = (const float*)d_in[4];
  const float* ew2 = (const float*)d_in[5];  const float* eb2 = (const float*)d_in[6];
  const float* pw1 = (const float*)d_in[7];  const float* pb1 = (const float*)d_in[8];
  const float* pw2 = (const float*)d_in[9];  const float* pb2 = (const float*)d_in[10];
  const float* gw1 = (const float*)d_in[11]; const float* gb1 = (const float*)d_in[12];
  const float* gw2 = (const float*)d_in[13]; const float* gb2 = (const float*)d_in[14];
  const float* bw1 = (const float*)d_in[15]; const float* bb1 = (const float*)d_in[16];
  const float* bw2 = (const float*)d_in[17]; const float* bb2 = (const float*)d_in[18];

  float* out       = (float*)d_out;
  float* xnew_sums = out;                       /* N*H: sums, then x_new  */
  float* edge_out  = out + (long)NNODES * H;    /* E*H: new_edge_attr     */
  float* counts    = (float*)d_ws;              /* N floats               */
  unsigned short* frags = (unsigned short*)((float*)d_ws + NNODES);

  zero_kernel<<<(NNODES * H) / 4 / 256, 256, 0, stream>>>(xnew_sums, counts);
  wtrans_kernel<<<(NFRAGS * 512) / 256, 256, 0, stream>>>(ew1, ew2, pw1, pw2,
                                                          gw1, gw2, bw1, bw2, frags);
  edge_kernel<<<(ETILES + 7) / 8, 256, 0, stream>>>(x, ei, ea, eb1, eb2, pb1, pb2,
                                                    frags, xnew_sums, counts, edge_out);
  node_kernel<<<(NTILES + 7) / 8, 256, 0, stream>>>(x, counts, gb1, gb2, bb1, bb2,
                                                    frags, xnew_sums);
}
